// MambaBlock_60327110639692
// MI455X (gfx1250) — matrix operand order repaired and checked by reading, not compiled, not run on hardware
//
#include <hip/hip_runtime.h>
#include <hip/hip_bf16.h>
#include <math.h>

// ---------------- Mamba2 block, MI455X (gfx1250): WMMA bf16 + TDM + async LDS ----------------
#define D_MODEL   768
#define D_INNER   1536
#define D_STATE   128
#define NHEADS    24
#define HEADDIM   64
#define CONV_DIM  1792
#define D_IN_PROJ 3352
#define BB        4
#define LL        2048
#define M_TOK     (BB * LL)          // 8192 tokens
#define EPS       1e-5f

typedef __bf16 bf16;
typedef __attribute__((ext_vector_type(16))) __bf16 v16bf;
typedef __attribute__((ext_vector_type(8)))  __bf16 bf16x8;
typedef __attribute__((ext_vector_type(8)))  float  v8f;
typedef unsigned int u32x4 __attribute__((ext_vector_type(4)));
typedef int          i32x4 __attribute__((ext_vector_type(4)));
typedef int          i32x8 __attribute__((ext_vector_type(8)));

// LDS row stride in bf16 elements (80 bytes: 16B aligned, 64-bank-friendly;
// TDM pad: 4 DWORDs after every 16 DWORDs reproduces this stride)
#define LDSTR 40

// ---------------- CDNA5 async global->LDS helpers (ASYNCcnt path) ----------------
__device__ __forceinline__ void async_ld_b128(void* lds, const void* g) {
    unsigned l = (unsigned)(unsigned long long)lds;
    asm volatile("global_load_async_to_lds_b128 %0, %1, off"
                 :: "v"(l), "v"((unsigned long long)g) : "memory");
}
__device__ __forceinline__ void async_ld_b32(void* lds, const void* g) {
    unsigned l = (unsigned)(unsigned long long)lds;
    asm volatile("global_load_async_to_lds_b32 %0, %1, off"
                 :: "v"(l), "v"((unsigned long long)g) : "memory");
}
__device__ __forceinline__ void wait_async0() {
    asm volatile("s_wait_asynccnt 0" ::: "memory");
}

// ---------------- CDNA5 Tensor Data Mover: 2D tile global->LDS ----------------
// D# per CDNA5 ISA. data_size_log: 0=1B,1=2B,2=4B.
// pad (optional): pad_int code (interval = 2<<code DWORDs), pad_amt code (code+1 DWORDs).
__device__ __forceinline__ void tdm_load_2d(unsigned lds_addr, const void* gaddr,
                                            unsigned tensor_d0, unsigned tensor_d1,
                                            unsigned tile_d0, unsigned tile_d1,
                                            unsigned stride_elems, unsigned data_size_log,
                                            unsigned pad_en, unsigned pad_int, unsigned pad_amt) {
    unsigned long long ga = (unsigned long long)gaddr;
    u32x4 g0;
    g0.x = 1u;                                             // count=1, user mode
    g0.y = lds_addr;                                       // LDS byte address
    g0.z = (unsigned)(ga & 0xFFFFFFFFu);                   // global_addr[31:0]
    g0.w = (unsigned)((ga >> 32) & 0x1FFFFFFu) | (2u << 30); // global_addr[56:32] | type=2
    i32x8 g1;
    g1[0] = (int)((data_size_log << 16) | (pad_en << 20) | (pad_int << 22) | (pad_amt << 25));
    g1[1] = (int)((tensor_d0 & 0xFFFFu) << 16);            // tensor_dim0[15:0]
    g1[2] = (int)(((tensor_d0 >> 16) & 0xFFFFu) | ((tensor_d1 & 0xFFFFu) << 16));
    g1[3] = (int)(((tensor_d1 >> 16) & 0xFFFFu) | ((tile_d0 & 0xFFFFu) << 16));
    g1[4] = (int)(tile_d1 & 0xFFFFu);                      // tile_dim1; tile_dim2=0
    g1[5] = (int)stride_elems;                             // tensor_dim0_stride[31:0]
    g1[6] = 0;                                             // stride hi / dim1_stride lo
    g1[7] = 0;
    i32x4 gz = {0, 0, 0, 0};
#if __has_builtin(__builtin_amdgcn_tensor_load_to_lds)
#if __clang_major__ >= 23
    i32x8 gz8 = {0, 0, 0, 0, 0, 0, 0, 0};
    __builtin_amdgcn_tensor_load_to_lds(g0, g1, gz, gz, gz8, 0);
#else
    __builtin_amdgcn_tensor_load_to_lds(g0, g1, gz, gz, 0);
#endif
#endif
}

// ---------------- fp32 -> bf16 conversion ----------------
__global__ void cvt_bf16_kernel(const float* __restrict__ src, bf16* __restrict__ dst, int n) {
    int i = blockIdx.x * 256 + threadIdx.x;
    if (i < n) dst[i] = (bf16)src[i];
}

// ---------------- WMMA GEMM: C[M,N] = A[M,K] * Bw[N,K]^T (+resid) ----------------
// Block tile 64x128, 8 waves (2x4), each wave 32x32 via 2x2 v_wmma_f32_16x16x32_bf16.
// Tiles staged by the Tensor Data Mover (TENSORcnt), fragments via ds_load_b128.
template<int K, int NREAL, bool RESID>
__global__ __launch_bounds__(256) void wmma_gemm_kernel(const bf16* __restrict__ A,
                                                        const bf16* __restrict__ Bw,
                                                        const float* __restrict__ resid,
                                                        float* __restrict__ C) {
    __shared__ bf16 As[64 * LDSTR];
    __shared__ bf16 Bs[128 * LDSTR];

    const int t    = threadIdx.x;
    const int m0   = blockIdx.x * 64;
    const int n0   = blockIdx.y * 128;
    const int w    = t >> 5;
    const int lane = t & 31;
    const int wm   = w >> 2;        // 0..1
    const int wn   = w & 3;         // 0..3
    const int lhi  = lane >> 4;     // 0..1
    const int llo  = lane & 15;     // 0..15

    const unsigned ldsA = (unsigned)(unsigned long long)(void*)As;
    const unsigned ldsB = (unsigned)(unsigned long long)(void*)Bs;
    const unsigned browsLeft = (unsigned)(NREAL - n0);
    const unsigned brows = browsLeft < 128u ? browsLeft : 128u;  // OOB rows -> zero via TDM

    v8f zero = {};
    v8f acc[2][2];
    acc[0][0] = zero; acc[0][1] = zero; acc[1][0] = zero; acc[1][1] = zero;

    for (int kc = 0; kc < K; kc += 32) {
        __syncthreads();   // previous iteration's fragment ds_loads complete in all waves
        if (w == 0) {
            // A tile: 64 rows x 32 bf16, row stride K elems, LDS pad -> 80B rows
            tdm_load_2d(ldsA, &A[(size_t)m0 * K + kc], /*td0*/K, /*td1*/64,
                        /*tile*/32, 64, /*stride*/K, /*2B*/1, /*pad*/1, 3, 3);
            // B tile: up to 128 rows x 32 bf16 (rows >= brows read as zero)
            tdm_load_2d(ldsB, &Bw[(size_t)n0 * K + kc], /*td0*/K, /*td1*/brows,
                        /*tile*/32, 128, /*stride*/K, /*2B*/1, /*pad*/1, 3, 3);
            __builtin_amdgcn_s_wait_tensorcnt(0);
        }
        if (kc + 32 < K)   // pull next A tile toward L2 while TDM runs
            __builtin_prefetch(&A[(size_t)(m0 + (t >> 2)) * K + kc + 32 + (t & 3) * 8], 0, 0);
        __syncthreads();

        // ---- fragments per documented gfx1250 WMMA VGPR layouts ----
        v16bf af[2], bf[2];
        #pragma unroll
        for (int tm = 0; tm < 2; tm++) {
            int row = wm * 32 + tm * 16 + llo;          // M = lane%16
            bf16x8 lo = *(const bf16x8*)&As[row * LDSTR + lhi * 8];        // K 0..7 / 8..15
            bf16x8 hi = *(const bf16x8*)&As[row * LDSTR + 16 + lhi * 8];   // K 16..23 / 24..31
            af[tm] = __builtin_shufflevector(lo, hi, 0,1,2,3,4,5,6,7,8,9,10,11,12,13,14,15);
        }
        #pragma unroll
        for (int tn = 0; tn < 2; tn++) {
            int row = wn * 32 + tn * 16 + llo;          // N = lane%16
            bf16x8 lo = *(const bf16x8*)&Bs[row * LDSTR + lhi * 8];        // K 0..7 / 8..15
            bf16x8 hi = *(const bf16x8*)&Bs[row * LDSTR + 16 + lhi * 8];   // K 16..23 / 24..31
            bf[tn] = __builtin_shufflevector(lo, hi, 0,1,2,3,4,5,6,7,8,9,10,11,12,13,14,15);
        }
        #pragma unroll
        for (int tm = 0; tm < 2; tm++)
            #pragma unroll
            for (int tn = 0; tn < 2; tn++)
                acc[tm][tn] = __builtin_amdgcn_wmma_f32_16x16x32_bf16(
                    false, af[tm], false, bf[tn], (short)0, acc[tm][tn], false, false);
    }

    // ---- store: C/D layout: row = 8*(lane/16)+v, col = lane%16 ----
    #pragma unroll
    for (int tm = 0; tm < 2; tm++)
        #pragma unroll
        for (int tn = 0; tn < 2; tn++) {
            int n = n0 + wn * 32 + tn * 16 + llo;
            if (n < NREAL) {
                #pragma unroll
                for (int v = 0; v < 8; v++) {
                    int m = m0 + wm * 32 + tm * 16 + lhi * 8 + v;
                    float val = acc[tm][tn][v];
                    if (RESID) val += resid[(size_t)m * NREAL + n];
                    C[(size_t)m * NREAL + n] = val;
                }
            }
        }
}

// ---------------- causal depthwise conv(4) + SiLU over xBC channels ----------------
__global__ __launch_bounds__(256) void conv_silu_kernel(const float* __restrict__ zxbcdt,
                                                        const float* __restrict__ conv_w,
                                                        const float* __restrict__ conv_b,
                                                        float* __restrict__ xBC) {
    int c = blockIdx.x * 256 + threadIdx.x;   // 0..1791
    int b = blockIdx.y;
    float w0 = conv_w[c * 4 + 0], w1 = conv_w[c * 4 + 1];
    float w2 = conv_w[c * 4 + 2], w3 = conv_w[c * 4 + 3];
    float bias = conv_b[c];
    float xm1 = 0.f, xm2 = 0.f, xm3 = 0.f;
    const float* src = zxbcdt + (size_t)b * LL * D_IN_PROJ + D_INNER + c;
    float* dst = xBC + (size_t)b * LL * CONV_DIM + c;
    for (int l = 0; l < LL; l++) {
        float cur = src[(size_t)l * D_IN_PROJ];
        float a = bias + w3 * cur + w2 * xm1 + w1 * xm2 + w0 * xm3;
        xm3 = xm2; xm2 = xm1; xm1 = cur;
        dst[(size_t)l * CONV_DIM] = a / (1.f + __expf(-a));
    }
}

// ---------------- dt = softplus(dt_raw + bias); dA = exp(dt * -exp(A_log)) ----------------
__global__ void dt_kernel(const float* __restrict__ zxbcdt, const float* __restrict__ dt_bias,
                          const float* __restrict__ A_log, float* __restrict__ dtarr,
                          float* __restrict__ dAarr) {
    int idx = blockIdx.x * 256 + threadIdx.x;   // == bl*24 + h, exactly 196608
    int h  = idx % NHEADS;
    int bl = idx / NHEADS;
    float x = zxbcdt[(size_t)bl * D_IN_PROJ + (D_INNER + CONV_DIM) + h] + dt_bias[h];
    float sp = (x > 20.f) ? x : log1pf(__expf(x));
    float A = -__expf(A_log[h]);
    dtarr[idx] = sp;
    dAarr[idx] = __expf(sp * A);
}

// ---------------- selective scan: one block per (b, h, p-half); state in registers ----------------
// 256 threads: 8 threads per p-row (32 rows), 16 n-elements per thread. 8 timesteps staged in LDS
// via async global->LDS copies (ASYNCcnt).
__global__ __launch_bounds__(256) void scan_kernel(const float* __restrict__ xBC,
                                                   const float* __restrict__ dAarr,
                                                   const float* __restrict__ dtarr,
                                                   const float* __restrict__ Dv,
                                                   float* __restrict__ y) {
    __shared__ float Bsh[8][D_STATE];
    __shared__ float Csh[8][D_STATE];
    __shared__ float xsh[8][32];
    __shared__ float dAsh[8], dtsh[8];

    int blk = blockIdx.x;
    int b   = blk / (NHEADS * 2);
    int rem = blk % (NHEADS * 2);
    int h   = rem >> 1;
    int p0  = (rem & 1) * 32;
    int t   = threadIdx.x;
    int pl  = t >> 3;            // local p row 0..31
    int nl  = (t & 7) * 16;      // n base, 16 per thread
    float Dh = Dv[h];

    float st[16];
    #pragma unroll
    for (int i = 0; i < 16; i++) st[i] = 0.f;

    const size_t rowB = (size_t)b * LL;
    const int lload = t >> 5;          // timestep this thread stages (0..7)
    const int eload = t & 31;
    for (int l0 = 0; l0 < LL; l0 += 8) {
        __syncthreads();
        {
            size_t base = (rowB + l0 + lload) * (size_t)CONV_DIM;
            int n4 = eload * 4;
            async_ld_b128(&Bsh[lload][n4], &xBC[base + D_INNER + n4]);
            async_ld_b128(&Csh[lload][n4], &xBC[base + D_INNER + D_STATE + n4]);
            async_ld_b32(&xsh[lload][eload], &xBC[base + h * HEADDIM + p0 + eload]);
        }
        if (t < 8)            dAsh[t]     = dAarr[(rowB + l0 + t) * NHEADS + h];
        else if (t < 16)      dtsh[t - 8] = dtarr[(rowB + l0 + t - 8) * NHEADS + h];
        wait_async0();
        __syncthreads();

        #pragma unroll
        for (int ll = 0; ll < 8; ll++) {
            float dA_t  = dAsh[ll];
            float coeff = dtsh[ll] * xsh[ll][pl];
            float yp = 0.f;
            #pragma unroll
            for (int i = 0; i < 16; i++) {
                st[i] = fmaf(st[i], dA_t, coeff * Bsh[ll][nl + i]);
                yp = fmaf(st[i], Csh[ll][nl + i], yp);
            }
            // reduce across the 8 lanes sharing this p (lane groups aligned within wave32)
            yp += __shfl_xor(yp, 4, 32);
            yp += __shfl_xor(yp, 2, 32);
            yp += __shfl_xor(yp, 1, 32);
            if ((t & 7) == 0)
                y[(rowB + l0 + ll) * (size_t)D_INNER + h * HEADDIM + p0 + pl]
                    = yp + Dh * xsh[ll][pl];
        }
    }
}

// ---------------- gated RMSNorm: y *= silu(z); rms-normalize; *w; emit bf16 ----------------
__global__ __launch_bounds__(256) void gated_norm_kernel(const float* __restrict__ y,
                                                         const float* __restrict__ zxbcdt,
                                                         const float* __restrict__ norm_w,
                                                         bf16* __restrict__ yb) {
    __shared__ float red[8];
    int r = blockIdx.x;
    int t = threadIdx.x;
    float g[6];
    float ss = 0.f;
    #pragma unroll
    for (int i = 0; i < 6; i++) {
        int idx = t + i * 256;
        float z = zxbcdt[(size_t)r * D_IN_PROJ + idx];
        float v = y[(size_t)r * D_INNER + idx] * (z / (1.f + __expf(-z)));
        g[i] = v;
        ss += v * v;
    }
    #pragma unroll
    for (int off = 16; off > 0; off >>= 1) ss += __shfl_xor(ss, off, 32);
    if ((t & 31) == 0) red[t >> 5] = ss;
    __syncthreads();
    if (t == 0) {
        float tot = 0.f;
        #pragma unroll
        for (int i = 0; i < 8; i++) tot += red[i];
        red[0] = rsqrtf(tot / (float)D_INNER + EPS);
    }
    __syncthreads();
    float rinv = red[0];
    #pragma unroll
    for (int i = 0; i < 6; i++) {
        int idx = t + i * 256;
        yb[(size_t)r * D_INNER + idx] = (bf16)(g[i] * rinv * norm_w[idx]);
    }
}

// ---------------- host launch ----------------
static inline size_t alignup(size_t v) { return (v + 255) & ~(size_t)255; }

extern "C" void kernel_launch(void* const* d_in, const int* in_sizes, int n_in,
                              void* d_out, int out_size, void* d_ws, size_t ws_size,
                              hipStream_t stream) {
    const float* x        = (const float*)d_in[0];
    const float* in_w     = (const float*)d_in[1];
    const float* conv_w   = (const float*)d_in[2];
    const float* conv_b   = (const float*)d_in[3];
    const float* dt_bias  = (const float*)d_in[4];
    const float* A_log    = (const float*)d_in[5];
    const float* Dv       = (const float*)d_in[6];
    const float* norm_w   = (const float*)d_in[7];
    const float* out_w    = (const float*)d_in[8];
    float* out            = (float*)d_out;

    // workspace layout
    char* p = (char*)d_ws;
    float* zxbcdt = (float*)p; p += alignup((size_t)M_TOK * D_IN_PROJ * 4);
    float* xBCc   = (float*)p; p += alignup((size_t)M_TOK * CONV_DIM * 4);
    float* dtarr  = (float*)p; p += alignup((size_t)M_TOK * NHEADS * 4);
    float* dAarr  = (float*)p; p += alignup((size_t)M_TOK * NHEADS * 4);
    float* yraw   = (float*)p; p += alignup((size_t)M_TOK * D_INNER * 4);
    bf16*  xb     = (bf16*)p;  p += alignup((size_t)M_TOK * D_MODEL * 2);
    bf16*  wib    = (bf16*)p;  p += alignup((size_t)D_IN_PROJ * D_MODEL * 2);
    bf16*  wob    = (bf16*)p;  p += alignup((size_t)D_MODEL * D_INNER * 2);
    bf16*  yb     = (bf16*)p;  p += alignup((size_t)M_TOK * D_INNER * 2);

    // 1) bf16 conversions for WMMA operands
    cvt_bf16_kernel<<<(M_TOK * D_MODEL + 255) / 256, 256, 0, stream>>>(x, xb, M_TOK * D_MODEL);
    cvt_bf16_kernel<<<(D_IN_PROJ * D_MODEL + 255) / 256, 256, 0, stream>>>(in_w, wib, D_IN_PROJ * D_MODEL);
    cvt_bf16_kernel<<<(D_MODEL * D_INNER + 255) / 256, 256, 0, stream>>>(out_w, wob, D_MODEL * D_INNER);

    // 2) in_proj GEMM (8192 x 3352, K=768), fp32 accumulate
    wmma_gemm_kernel<D_MODEL, D_IN_PROJ, false>
        <<<dim3(M_TOK / 64, (D_IN_PROJ + 127) / 128), 256, 0, stream>>>(xb, wib, nullptr, zxbcdt);

    // 3) depthwise conv + SiLU
    conv_silu_kernel<<<dim3(CONV_DIM / 256, BB), 256, 0, stream>>>(zxbcdt, conv_w, conv_b, xBCc);

    // 4) dt / dA scalars
    dt_kernel<<<(M_TOK * NHEADS) / 256, 256, 0, stream>>>(zxbcdt, dt_bias, A_log, dtarr, dAarr);

    // 5) selective scan (192 blocks: b x h x p-half), state resident in VGPRs
    scan_kernel<<<BB * NHEADS * 2, 256, 0, stream>>>(xBCc, dAarr, dtarr, Dv, yraw);

    // 6) gated RMSNorm -> bf16
    gated_norm_kernel<<<M_TOK, 256, 0, stream>>>(yraw, zxbcdt, norm_w, yb);

    // 7) out_proj GEMM (8192 x 768, K=1536) + residual, write fp32 output
    wmma_gemm_kernel<D_INNER, D_MODEL, true>
        <<<dim3(M_TOK / 64, D_MODEL / 128), 256, 0, stream>>>(yb, wob, x, out);
}
